// GAT_20710332301836
// MI455X (gfx1250) — matrix operand order repaired and checked by reading, not compiled, not run on hardware
//
#include <hip/hip_runtime.h>
#include <hip/hip_bf16.h>
#include <stdint.h>

// ---------------------------------------------------------------------------
// Types for CDNA5 WMMA (wave32)
// ---------------------------------------------------------------------------
typedef __attribute__((ext_vector_type(16))) __bf16        v16bf;
typedef __attribute__((ext_vector_type(8)))  float         v8f;
typedef __attribute__((ext_vector_type(8)))  unsigned int  v8u;

#define SLOPE 0.2f
#define EPS_BN 1e-5f

// ---------------------------------------------------------------------------
// Helpers
// ---------------------------------------------------------------------------
__device__ __forceinline__ unsigned int f2bf(float f) {
    unsigned int u = __float_as_uint(f);
    u += 0x7fffu + ((u >> 16) & 1u);        // round-to-nearest-even
    return (u >> 16) & 0xffffu;
}

__device__ __forceinline__ void atomicMaxF(float* addr, float val) {
    unsigned int* ua = (unsigned int*)addr;
    unsigned int old = *ua;
    while (__uint_as_float(old) < val) {
        unsigned int assumed = old;
        old = atomicCAS(ua, assumed, __float_as_uint(val));
        if (old == assumed) break;
    }
}

// ---------------------------------------------------------------------------
// Pack A (row-major fp32, rows = mtiles*16, ld columns) into the wave32 bf16
// 16x32 A-fragment layout: lane holds row = lane&15;
//   k0 = 2*v + (v>=4 ? 8:0) + (lane>=16 ? 8:0), pair (k0, k0+1) per dword.
// Storage: out[(((mt*ktiles)+kt)*32 + lane)*8 + v]  -> per-lane 8 dwords contiguous
// ---------------------------------------------------------------------------
__global__ void pack_a_bf16(const float* __restrict__ X, unsigned int* __restrict__ out,
                            int mtiles, int ktiles, int ld) {
    int idx = blockIdx.x * blockDim.x + threadIdx.x;
    int total = mtiles * ktiles * 256;
    if (idx >= total) return;
    int v    = idx & 7;
    int t    = idx >> 3;
    int lane = t & 31;  t >>= 5;
    int kt   = t % ktiles;
    int mt   = t / ktiles;
    int row  = mt * 16 + (lane & 15);
    int k0   = kt * 32 + 2 * v + (v >= 4 ? 8 : 0) + (lane >= 16 ? 8 : 0);
    const float* r = X + (size_t)row * ld + k0;
    out[idx] = f2bf(r[0]) | (f2bf(r[1]) << 16);
}

// ---------------------------------------------------------------------------
// Pack B (row-major fp32 W[k][col], `cols` real columns, K = ktiles*32) into
// wave32 bf16 32x16 B-fragment layout: lane col = lane&15,
//   k = 2*v + (v>=4 ? 8:0) + (lane>=16 ? 8:0), pair (k, k+1). Zero-pad cols >= `cols`.
// ---------------------------------------------------------------------------
__global__ void pack_b_bf16(const float* __restrict__ W, unsigned int* __restrict__ out,
                            int ntiles, int ktiles, int cols) {
    int idx = blockIdx.x * blockDim.x + threadIdx.x;
    int total = ntiles * ktiles * 256;
    if (idx >= total) return;
    int v    = idx & 7;
    int t    = idx >> 3;
    int lane = t & 31;  t >>= 5;
    int kt   = t % ktiles;
    int nt   = t / ktiles;
    int col  = nt * 16 + (lane & 15);
    int k    = kt * 32 + 2 * v + (v >= 4 ? 8 : 0) + (lane >= 16 ? 8 : 0);
    float f0 = 0.f, f1 = 0.f;
    if (col < cols) {
        f0 = W[(size_t)k * cols + col];
        f1 = W[(size_t)(k + 1) * cols + col];
    }
    out[idx] = f2bf(f0) | (f2bf(f1) << 16);
}

// ---------------------------------------------------------------------------
// Tiled GEMM via v_wmma_f32_16x16x32_bf16. One wave per 16x16 output tile.
// Early-exit condition is wave-uniform so EXEC is all-ones at every WMMA.
// ---------------------------------------------------------------------------
__global__ void gemm_wmma_bf16(const unsigned int* __restrict__ A,
                               const unsigned int* __restrict__ B,
                               float* __restrict__ C,
                               int mtiles, int ntiles, int ktiles,
                               int ldc, int realCols) {
    int wave = (blockIdx.x * blockDim.x + threadIdx.x) >> 5;
    int lane = threadIdx.x & 31;
    if (wave >= mtiles * ntiles) return;            // wave-uniform
    int mt = wave / ntiles;
    int nt = wave % ntiles;

    const unsigned int* ap = A + ((size_t)(mt * ktiles) * 32 + lane) * 8;
    const unsigned int* bp = B + ((size_t)(nt * ktiles) * 32 + lane) * 8;

    v8f acc = {0.f, 0.f, 0.f, 0.f, 0.f, 0.f, 0.f, 0.f};
    for (int kt = 0; kt < ktiles; ++kt) {
        uint4 a0 = *(const uint4*)(ap);
        uint4 a1 = *(const uint4*)(ap + 4);
        uint4 b0 = *(const uint4*)(bp);
        uint4 b1 = *(const uint4*)(bp + 4);
        __builtin_prefetch(ap + 256, 0, 0);         // global_prefetch_b8 next A frag
        ap += 256; bp += 256;
        v8u au = {a0.x, a0.y, a0.z, a0.w, a1.x, a1.y, a1.z, a1.w};
        v8u bu = {b0.x, b0.y, b0.z, b0.w, b1.x, b1.y, b1.z, b1.w};
        acc = __builtin_amdgcn_wmma_f32_16x16x32_bf16(
                false, __builtin_bit_cast(v16bf, au),
                false, __builtin_bit_cast(v16bf, bu),
                (short)0, acc, false, false);
    }
    // C/D layout: vgpr r, lanes 0-15 -> M=r, lanes 16-31 -> M=8+r; N = lane&15
    int row0 = mt * 16 + ((lane >> 4) << 3);
    int col  = nt * 16 + (lane & 15);
    if (col < realCols) {
        for (int r = 0; r < 8; ++r)
            C[(size_t)(row0 + r) * ldc + col] = acc[r];
    }
}

// ---------------------------------------------------------------------------
// el[n,h] = <ft[n,h,:], al[h,:]> ; er likewise
// ---------------------------------------------------------------------------
__global__ void attn_scores(const float* __restrict__ ft,
                            const float* __restrict__ al, const float* __restrict__ ar,
                            float* __restrict__ el, float* __restrict__ er,
                            int n, int H, int D) {
    int i = blockIdx.x * blockDim.x + threadIdx.x;
    if (i >= n * H) return;
    int h = i % H;
    const float* f = ft + (size_t)i * D;
    float sl = 0.f, sr = 0.f;
    for (int d = 0; d < D; ++d) {
        float v = f[d];
        sl += v * al[h * D + d];
        sr += v * ar[h * D + d];
    }
    el[i] = sl; er[i] = sr;
}

__global__ void fill_f32(float* __restrict__ p, float v, int n) {
    int i = blockIdx.x * blockDim.x + threadIdx.x;
    if (i < n) p[i] = v;
}

// segment-max of leaky_relu(el[src]+er[dst]) per (dst, head)
__global__ void edge_max(const int* __restrict__ src, const int* __restrict__ dst,
                         const float* __restrict__ el, const float* __restrict__ er,
                         float* __restrict__ m, int nE, int H) {
    int i = blockIdx.x * blockDim.x + threadIdx.x;
    if (i >= nE * H) return;
    int e = i / H, h = i % H;
    int sn = src[e], dn = dst[e];
    float v = el[sn * H + h] + er[dn * H + h];
    v = v > 0.f ? v : SLOPE * v;
    atomicMaxF(&m[dn * H + h], v);
}

// ex = exp(e - m[dst]); s[dst] += ex; stash ex per edge
__global__ void edge_exp(const int* __restrict__ src, const int* __restrict__ dst,
                         const float* __restrict__ el, const float* __restrict__ er,
                         const float* __restrict__ m, float* __restrict__ s,
                         float* __restrict__ exv, int nE, int H) {
    int i = blockIdx.x * blockDim.x + threadIdx.x;
    if (i >= nE * H) return;
    int e = i / H, h = i % H;
    int sn = src[e], dn = dst[e];
    float v = el[sn * H + h] + er[dn * H + h];
    v = v > 0.f ? v : SLOPE * v;
    float mm = m[dn * H + h];
    if (mm < -3.0e38f) mm = 0.f;                    // isfinite() guard
    float ex = expf(v - mm);
    exv[i] = ex;
    atomicAdd(&s[dn * H + h], ex);
}

// wave32 per edge: out[dst, :] += (ex/s[dst]) * ft[src, :]
__global__ void edge_aggregate(const int* __restrict__ src, const int* __restrict__ dst,
                               const float* __restrict__ exv, const float* __restrict__ s,
                               const float* __restrict__ ft, float* __restrict__ out,
                               int nE, int H, int D) {
    int wave = (blockIdx.x * blockDim.x + threadIdx.x) >> 5;
    int lane = threadIdx.x & 31;
    if (wave >= nE) return;
    int sn = src[wave], dn = dst[wave];
    int HD = H * D;
    for (int hd = lane; hd < HD; hd += 32) {
        int h = hd / D;
        float a = exv[(size_t)wave * H + h] / fmaxf(s[dn * H + h], 1e-9f);
        atomicAdd(&out[(size_t)dn * HD + hd], a * ft[(size_t)sn * HD + hd]);
    }
}

// per-channel partial sums (coalesced: thread = channel), atomics to global
__global__ void bn_stats(const float* __restrict__ h, float* __restrict__ sum,
                         float* __restrict__ sumsq, int n, int ch, int rowsPerBlock) {
    int c  = threadIdx.x;                            // blockDim.x == ch
    int r0 = blockIdx.x * rowsPerBlock;
    int r1 = r0 + rowsPerBlock; if (r1 > n) r1 = n;
    float s = 0.f, ss = 0.f;
    for (int r = r0; r < r1; ++r) {
        float v = h[(size_t)r * ch + c];
        s += v; ss += v * v;
    }
    atomicAdd(&sum[c], s);
    atomicAdd(&sumsq[c], ss);
}

// BN(train-stats) + ELU + repack into bf16 A-fragment layout for GEMM2
__global__ void bn_elu_pack(const float* __restrict__ hin,
                            const float* __restrict__ sum, const float* __restrict__ sumsq,
                            const float* __restrict__ gamma, const float* __restrict__ beta,
                            unsigned int* __restrict__ out,
                            int mtiles, int ktiles, float invN) {
    int idx = blockIdx.x * blockDim.x + threadIdx.x;
    int total = mtiles * ktiles * 256;
    if (idx >= total) return;
    int v    = idx & 7;
    int t    = idx >> 3;
    int lane = t & 31;  t >>= 5;
    int kt   = t % ktiles;
    int mt   = t / ktiles;
    int row  = mt * 16 + (lane & 15);
    int ch   = ktiles * 32;
    int c0   = kt * 32 + 2 * v + (v >= 4 ? 8 : 0) + (lane >= 16 ? 8 : 0);

    unsigned int pk = 0;
    for (int j = 0; j < 2; ++j) {
        int c = c0 + j;
        float x  = hin[(size_t)row * ch + c];
        float mu = sum[c] * invN;
        float vr = sumsq[c] * invN - mu * mu;
        float xn = (x - mu) * rsqrtf(vr + EPS_BN) * gamma[c] + beta[c];
        xn = xn > 0.f ? xn : (expf(xn) - 1.f);       // ELU
        pk |= f2bf(xn) << (16 * j);
    }
    out[idx] = pk;
}

__global__ void log_softmax_k(const float* __restrict__ in, float* __restrict__ out,
                              int n, int c) {
    int i = blockIdx.x * blockDim.x + threadIdx.x;
    if (i >= n) return;
    const float* r = in + (size_t)i * c;
    float mx = r[0];
    for (int j = 1; j < c; ++j) mx = fmaxf(mx, r[j]);
    float s = 0.f;
    for (int j = 0; j < c; ++j) s += expf(r[j] - mx);
    float ls = logf(s);
    float* o = out + (size_t)i * c;
    for (int j = 0; j < c; ++j) o[j] = r[j] - mx - ls;
}

// ---------------------------------------------------------------------------
// Host orchestration
// ---------------------------------------------------------------------------
extern "C" void kernel_launch(void* const* d_in, const int* in_sizes, int n_in,
                              void* d_out, int out_size, void* d_ws, size_t ws_size,
                              hipStream_t stream) {
    const float* x     = (const float*)d_in[0];
    const int*   src   = (const int*)  d_in[1];
    const int*   dst   = (const int*)  d_in[2];
    const float* W1    = (const float*)d_in[3];
    const float* al1   = (const float*)d_in[4];
    const float* ar1   = (const float*)d_in[5];
    const float* gamma = (const float*)d_in[6];
    const float* beta  = (const float*)d_in[7];
    const float* W2    = (const float*)d_in[8];
    const float* al2   = (const float*)d_in[9];
    const float* ar2   = (const float*)d_in[10];

    const int N  = 50000, IN = 256, H1 = 4, D1 = 32, HD1 = 128, C = 40;
    const int E  = in_sizes[1];                      // 850000
    const int MT = N / 16;                            // 3125 (exact)
    const int KT1 = IN / 32, NT1 = HD1 / 16;          // 8, 8
    const int KT2 = HD1 / 32, NT2 = 3;                // 4, 3 (48 cols padded)

    uint8_t* ws = (uint8_t*)d_ws;
    size_t off = 0;
    auto take = [&](size_t bytes) -> void* {
        void* p = ws + off;
        off += (bytes + 255) & ~(size_t)255;
        return p;
    };
    unsigned int* apack = (unsigned int*)take((size_t)MT * KT1 * 256 * 4); // reused L2
    unsigned int* wp1   = (unsigned int*)take((size_t)NT1 * KT1 * 256 * 4);
    unsigned int* wp2   = (unsigned int*)take((size_t)NT2 * KT2 * 256 * 4);
    float* ft1   = (float*)take((size_t)N * HD1 * 4);   // reused as logits
    float* hout  = (float*)take((size_t)N * HD1 * 4);   // reused as ft2
    float* exv   = (float*)take((size_t)E * H1 * 4);
    float* el    = (float*)take((size_t)N * H1 * 4);
    float* er    = (float*)take((size_t)N * H1 * 4);
    float* mbuf  = (float*)take((size_t)N * H1 * 4);
    float* sbuf  = (float*)take((size_t)N * H1 * 4);
    float* bnsum = (float*)take(HD1 * 4);
    float* bnsq  = (float*)take(HD1 * 4);
    float* ft2    = hout;
    float* logits = ft1;

    auto grid = [](long n, int b) { return (int)((n + b - 1) / b); };
    const float NEGINF = -__builtin_huge_valf();

    // ---- Layer 1 ----
    pack_a_bf16<<<grid((long)MT * KT1 * 256, 256), 256, 0, stream>>>(x, apack, MT, KT1, IN);
    pack_b_bf16<<<grid((long)NT1 * KT1 * 256, 256), 256, 0, stream>>>(W1, wp1, NT1, KT1, HD1);
    gemm_wmma_bf16<<<grid((long)MT * NT1 * 32, 256), 256, 0, stream>>>(
        apack, wp1, ft1, MT, NT1, KT1, HD1, HD1);

    attn_scores<<<grid((long)N * H1, 256), 256, 0, stream>>>(ft1, al1, ar1, el, er, N, H1, D1);

    fill_f32<<<grid((long)N * H1, 256), 256, 0, stream>>>(mbuf, NEGINF, N * H1);
    fill_f32<<<grid((long)N * H1, 256), 256, 0, stream>>>(sbuf, 0.f, N * H1);
    fill_f32<<<grid((long)N * HD1, 256), 256, 0, stream>>>(hout, 0.f, N * HD1);

    edge_max<<<grid((long)E * H1, 256), 256, 0, stream>>>(src, dst, el, er, mbuf, E, H1);
    edge_exp<<<grid((long)E * H1, 256), 256, 0, stream>>>(src, dst, el, er, mbuf, sbuf, exv, E, H1);
    edge_aggregate<<<grid((long)E * 32, 256), 256, 0, stream>>>(
        src, dst, exv, sbuf, ft1, hout, E, H1, D1);

    // ---- BatchNorm + ELU + repack ----
    fill_f32<<<1, 256, 0, stream>>>(bnsum, 0.f, HD1);
    fill_f32<<<1, 256, 0, stream>>>(bnsq, 0.f, HD1);
    bn_stats<<<grid(N, 256), HD1, 0, stream>>>(hout, bnsum, bnsq, N, HD1, 256);
    bn_elu_pack<<<grid((long)MT * KT2 * 256, 256), 256, 0, stream>>>(
        hout, bnsum, bnsq, gamma, beta, apack, MT, KT2, 1.0f / (float)N);

    // ---- Layer 2 ----
    pack_b_bf16<<<grid((long)NT2 * KT2 * 256, 256), 256, 0, stream>>>(W2, wp2, NT2, KT2, C);
    gemm_wmma_bf16<<<grid((long)MT * NT2 * 32, 256), 256, 0, stream>>>(
        apack, wp2, ft2, MT, NT2, KT2, C, C);

    attn_scores<<<grid(N, 256), 256, 0, stream>>>(ft2, al2, ar2, el, er, N, 1, C);

    fill_f32<<<grid(N, 256), 256, 0, stream>>>(mbuf, NEGINF, N);
    fill_f32<<<grid(N, 256), 256, 0, stream>>>(sbuf, 0.f, N);
    fill_f32<<<grid((long)N * C, 256), 256, 0, stream>>>(logits, 0.f, N * C);

    edge_max<<<grid(E, 256), 256, 0, stream>>>(src, dst, el, er, mbuf, E, 1);
    edge_exp<<<grid(E, 256), 256, 0, stream>>>(src, dst, el, er, mbuf, sbuf, exv, E, 1);
    edge_aggregate<<<grid((long)E * 32, 256), 256, 0, stream>>>(
        src, dst, exv, sbuf, ft2, logits, E, 1, C);

    log_softmax_k<<<grid(N, 256), 256, 0, stream>>>(logits, (float*)d_out, N, C);
}
